// InstanceAdaptiveKeypointDetector2_79714593014463
// MI455X (gfx1250) — matrix operand order repaired and checked by reading, not compiled, not run on hardware
//
#include <hip/hip_runtime.h>
#include <hip/hip_bf16.h>

// ---------------- problem constants ----------------
constexpr int Bn = 8;        // batch
constexpr int Np = 16384;    // points
constexpr int Cc = 256;      // channels
constexpr int C2 = 512;      // layer-3 width
constexpr int Kk = 16;       // keypoints
constexpr int KO = 48;       // K*3 offset channels

// ---------------- workspace layout (bytes) ----------------
constexpr size_t OFF_VOTED = 0;
constexpr size_t SZ_VOTED  = (size_t)Bn * Np * Kk * 3 * 4;     // 25,165,824
constexpr size_t OFF_W1H   = OFF_VOTED + SZ_VOTED;
constexpr size_t OFF_W2H   = OFF_W1H  + (size_t)65536 * 2;
constexpr size_t OFF_W3H   = OFF_W2H  + (size_t)65536 * 2;
constexpr size_t OFF_WOH   = OFF_W3H  + (size_t)131072 * 2;
constexpr size_t OFF_WM1H  = OFF_WOH  + (size_t)24576 * 2;
constexpr size_t OFF_WM2H  = OFF_WM1H + (size_t)65536 * 2;
constexpr size_t OFF_WM3H  = OFF_WM2H + (size_t)65536 * 2;
constexpr size_t OFF_S1    = OFF_WM3H + (size_t)65536 * 2;
constexpr size_t OFF_T1    = OFF_S1  + 256 * 4;
constexpr size_t OFF_S2    = OFF_T1  + 256 * 4;
constexpr size_t OFF_T2    = OFF_S2  + 256 * 4;
constexpr size_t OFF_S3    = OFF_T2  + 256 * 4;
constexpr size_t OFF_T3    = OFF_S3  + 512 * 4;
constexpr size_t OFF_SM1   = OFF_T3  + 512 * 4;
constexpr size_t OFF_TM1   = OFF_SM1 + 256 * 4;
constexpr size_t OFF_SM2   = OFF_TM1 + 256 * 4;
constexpr size_t OFF_TM2   = OFF_SM2 + 256 * 4;
constexpr size_t OFF_VSUM  = OFF_TM2 + 256 * 4;                // 384 f32
constexpr size_t OFF_SUMEXP= OFF_VSUM + 384 * 4;               // 128 f32
constexpr size_t OFF_KPPOS = OFF_SUMEXP + 128 * 4;             // 384 f32
constexpr size_t OFF_KPRAW = OFF_KPPOS + 384 * 4;              // 32768 f32

// ---------------- WMMA helpers (CDNA5, wave32) ----------------
typedef __attribute__((ext_vector_type(16))) _Float16 v16h;
typedef __attribute__((ext_vector_type(8)))  _Float16 v8h;
typedef __attribute__((ext_vector_type(4)))  _Float16 v4h;
typedef __attribute__((ext_vector_type(8)))  float    v8f;

// A operand, 16x32 f16: lane L holds row M=L%16; elem e = K hi*16+e
// (hi = L/16). rowPtr points at (row, k0).
__device__ __forceinline__ v16h wmma_ld_a(const _Float16* rowPtr, int hi) {
  v8h lo = *(const v8h*)(rowPtr + hi * 16);
  v8h hh = *(const v8h*)(rowPtr + hi * 16 + 8);
  v16h r;
#pragma unroll
  for (int i = 0; i < 8; ++i) { r[i] = lo[i]; r[i + 8] = hh[i]; }
  return r;
}

// B operand, 32x16 f16: lane L holds col N=L%16; elem e = K (L/16)*16+e.
// p points at (col-row-of-W, k0 + hi*16): 16 contiguous halves.
__device__ __forceinline__ v16h wmma_ld_b(const _Float16* p) {
  v8h lo = *(const v8h*)p;
  v8h hh = *(const v8h*)(p + 8);
  v16h r;
#pragma unroll
  for (int i = 0; i < 8; ++i) { r[i] = lo[i]; r[i + 8] = hh[i]; }
  return r;
}

__device__ __forceinline__ v8f wmma16(v16h a, v16h b, v8f c) {
  return __builtin_amdgcn_wmma_f32_16x16x32_f16(false, a, false, b, (short)0, c,
                                                false, false);
}

// ---------------- kernel 0: weight convert + BN fuse + zeroing ----------------
struct PrepArgs {
  const float *W1, *W2, *W3, *Wo, *Wm1, *Wm2, *Wm3;
  const float *g1, *b1, *m1, *v1;
  const float *g2, *b2, *m2, *v2;
  const float *g3, *b3, *m3, *v3;
  const float *gm1, *bm1, *mm1, *vm1;
  const float *gm2, *bm2, *mm2, *vm2;
  char* ws;
};

__global__ void k_prep(PrepArgs p) {
  const int i = blockIdx.x * 256 + threadIdx.x;
  char* ws = p.ws;
  _Float16* W1h  = (_Float16*)(ws + OFF_W1H);
  _Float16* W2h  = (_Float16*)(ws + OFF_W2H);
  _Float16* W3h  = (_Float16*)(ws + OFF_W3H);
  _Float16* Woh  = (_Float16*)(ws + OFF_WOH);
  _Float16* Wm1h = (_Float16*)(ws + OFF_WM1H);
  _Float16* Wm2h = (_Float16*)(ws + OFF_WM2H);
  _Float16* Wm3h = (_Float16*)(ws + OFF_WM3H);
  if (i < 65536) {
    W1h[i]  = (_Float16)p.W1[i];
    W2h[i]  = (_Float16)p.W2[i];
    Wm1h[i] = (_Float16)p.Wm1[i];
    Wm2h[i] = (_Float16)p.Wm2[i];
    Wm3h[i] = (_Float16)p.Wm3[i];
  }
  if (i < 131072) W3h[i] = (_Float16)p.W3[i];
  if (i < 24576)  Woh[i] = (_Float16)p.Wo[i];
  if (i < 256) {
    float s;
    s = p.g1[i] * rsqrtf(p.v1[i] + 1e-5f);
    ((float*)(ws + OFF_S1))[i] = s; ((float*)(ws + OFF_T1))[i] = p.b1[i] - p.m1[i] * s;
    s = p.g2[i] * rsqrtf(p.v2[i] + 1e-5f);
    ((float*)(ws + OFF_S2))[i] = s; ((float*)(ws + OFF_T2))[i] = p.b2[i] - p.m2[i] * s;
    s = p.gm1[i] * rsqrtf(p.vm1[i] + 1e-5f);
    ((float*)(ws + OFF_SM1))[i] = s; ((float*)(ws + OFF_TM1))[i] = p.bm1[i] - p.mm1[i] * s;
    s = p.gm2[i] * rsqrtf(p.vm2[i] + 1e-5f);
    ((float*)(ws + OFF_SM2))[i] = s; ((float*)(ws + OFF_TM2))[i] = p.bm2[i] - p.mm2[i] * s;
  }
  if (i < 512) {
    float s = p.g3[i] * rsqrtf(p.v3[i] + 1e-5f);
    ((float*)(ws + OFF_S3))[i] = s; ((float*)(ws + OFF_T3))[i] = p.b3[i] - p.m3[i] * s;
  }
  if (i < 384)   ((float*)(ws + OFF_VSUM))[i]   = 0.f;
  if (i < 128)   ((float*)(ws + OFF_SUMEXP))[i] = 0.f;
  if (i < 32768) ((float*)(ws + OFF_KPRAW))[i]  = 0.f;
}

// ---------------- kernel 1: backbone + offset head + vote accumulation ----------
constexpr int TM  = 128;   // rows per block (8 waves x 16)
constexpr int SA  = 264;   // activation LDS stride (halves), 528B rows -> 16B aligned
constexpr int SST = 40;    // stage stride (halves)
constexpr int SMEM_K1 = 2 * TM * SA * 2 + 8 * 16 * SST * 2 + TM * 3 * 4;

__global__ __launch_bounds__(256) void k_backbone(
    const float* __restrict__ ff, const float* __restrict__ xyz,
    const _Float16* __restrict__ W1h, const _Float16* __restrict__ W2h,
    const _Float16* __restrict__ W3h, const _Float16* __restrict__ Woh,
    const float* __restrict__ s1, const float* __restrict__ t1,
    const float* __restrict__ s2, const float* __restrict__ t2,
    const float* __restrict__ s3, const float* __restrict__ t3,
    const float* __restrict__ bo,
    float* __restrict__ voted, float* __restrict__ vote_sum) {
  extern __shared__ char smem[];
  _Float16* actA  = (_Float16*)smem;                 // TM x SA (ff, then x2)
  _Float16* actB  = actA + TM * SA;                  // TM x SA (x1)
  _Float16* stage = actB + TM * SA;                  // 8 waves x 16 x SST (x3 chunk)
  float*    xyzs  = (float*)(stage + 8 * 16 * SST);  // TM x 3

  const int blk = blockIdx.x;
  const int b   = blk / (Np / TM);
  const int n0  = (blk % (Np / TM)) * TM;
  const int tid = threadIdx.x;
  const int wave = tid >> 5, lane = tid & 31, l16 = lane & 15, hi = lane >> 4;

  // cooperative vectorized f32->f16 load of the feature tile + xyz tile
  const float* ffp = ff + ((size_t)b * Np + n0) * Cc;
  for (int i = tid; i < TM * Cc / 4; i += 256) {
    const int r = i >> 6, c4 = (i & 63) * 4;
    float4 f = ((const float4*)ffp)[i];
    v4h h = { (_Float16)f.x, (_Float16)f.y, (_Float16)f.z, (_Float16)f.w };
    *(v4h*)(actA + r * SA + c4) = h;
  }
  const float* xp = xyz + ((size_t)b * Np + n0) * 3;
  for (int i = tid; i < TM * 3; i += 256) xyzs[i] = xp[i];
  __syncthreads();

  const int m0 = wave * 16;  // this wave's private 16-row strip

  // ---- layers 1 and 2: 256 -> 256, BN+ReLU ----
  for (int layer = 0; layer < 2; ++layer) {
    const _Float16* src = layer ? actB : actA;
    _Float16*       dst = layer ? actA : actB;
    const _Float16* Wh  = layer ? W2h : W1h;
    const float*    ss  = layer ? s2 : s1;
    const float*    tt  = layer ? t2 : t1;
    const _Float16* arow = src + (m0 + l16) * SA;
    // hoist: this wave's entire A strip lives in registers for the layer
    v16h areg[8];
#pragma unroll
    for (int kc = 0; kc < 8; ++kc) areg[kc] = wmma_ld_a(arow + kc * 32, hi);
    for (int nt = 0; nt < 16; ++nt) {
      const int nc = nt * 16;
      if (nt < 15)  // prefetch next B tile row (global_prefetch_b8)
        __builtin_prefetch(Wh + (size_t)(nc + 16 + l16) * Cc + hi * 16, 0, 1);
      v8f acc = {};
#pragma unroll
      for (int kc = 0; kc < 8; ++kc) {
        v16h bm = wmma_ld_b(Wh + (size_t)(nc + l16) * Cc + kc * 32 + hi * 16);
        acc = wmma16(areg[kc], bm, acc);
      }
      const int col = nc + l16;
      const float s = ss[col], t = tt[col];
#pragma unroll
      for (int r = 0; r < 8; ++r) {
        float v = fmaxf(acc[r] * s + t, 0.f);
        dst[(m0 + r + hi * 8) * SA + col] = (_Float16)v;
      }
    }
  }

  // ---- layer 3 (256 -> 512, BN+ReLU) fused with offset head (512 -> 48) ----
  {
    const _Float16* arow = actA + (m0 + l16) * SA;
    _Float16* stg = stage + wave * 16 * SST;
    const _Float16* srow = stg + l16 * SST;
    v16h areg[8];
#pragma unroll
    for (int kc = 0; kc < 8; ++kc) areg[kc] = wmma_ld_a(arow + kc * 32, hi);
    v8f oacc0 = {}, oacc1 = {}, oacc2 = {};
    for (int ch = 0; ch < 16; ++ch) {  // 32 layer-3 channels at a time
#pragma unroll
      for (int sub = 0; sub < 2; ++sub) {
        const int d3 = ch * 32 + sub * 16;
        v8f acc = {};
#pragma unroll
        for (int kc = 0; kc < 8; ++kc) {
          v16h bm = wmma_ld_b(W3h + (size_t)(d3 + l16) * Cc + kc * 32 + hi * 16);
          acc = wmma16(areg[kc], bm, acc);
        }
        const int col = d3 + l16;
        const float s = s3[col], t = t3[col];
#pragma unroll
        for (int r = 0; r < 8; ++r) {
          float v = fmaxf(acc[r] * s + t, 0.f);
          stg[(r + hi * 8) * SST + sub * 16 + l16] = (_Float16)v;
        }
      }
      // multiply the 16x32 x3 chunk into the 48-wide offset accumulators
      v16h a = wmma_ld_a(srow, hi);
      v16h bm;
      bm = wmma_ld_b(Woh + (size_t)(0 * 16 + l16) * C2 + ch * 32 + hi * 16);
      oacc0 = wmma16(a, bm, oacc0);
      bm = wmma_ld_b(Woh + (size_t)(1 * 16 + l16) * C2 + ch * 32 + hi * 16);
      oacc1 = wmma16(a, bm, oacc1);
      bm = wmma_ld_b(Woh + (size_t)(2 * 16 + l16) * C2 + ch * 32 + hi * 16);
      oacc2 = wmma16(a, bm, oacc2);
    }
    // epilogue: voted = xyz + offsets + bo; store + per-(b,k,coord) sums
    v8f oaccs[3] = {oacc0, oacc1, oacc2};
#pragma unroll
    for (int ot = 0; ot < 3; ++ot) {
      const int col = ot * 16 + l16;            // 0..47 = k*3+coord
      const int kidx = col / 3, coord = col % 3;
      const float bov = bo[col];
      float ps = 0.f;
#pragma unroll
      for (int r = 0; r < 8; ++r) {
        const int m = m0 + r + hi * 8;
        float v = oaccs[ot][r] + bov + xyzs[m * 3 + coord];
        voted[((size_t)((size_t)b * Np + n0 + m) * Kk + kidx) * 3 + coord] = v;
        ps += v;
      }
      ps += __shfl_xor(ps, 16, 32);             // fold the two lane-halves
      if (hi == 0) atomicAdd(&vote_sum[b * KO + col], ps);
    }
  }
}

// ---------------- kernel 2: kp_pos = vote_sum / N ----------------
__global__ void k_kppos(const float* __restrict__ vsum,
                        float* __restrict__ kppos_ws, float* __restrict__ out) {
  const int i = threadIdx.x;
  if (i < 384) {
    float v = vsum[i] * (1.f / (float)Np);
    kppos_ws[i] = v;
    out[i] = v;
  }
}

// ---------------- kernel 3: exp-weighted pooling, kp_raw = wexp^T @ feat -------
constexpr int NCH = 8;            // chunks per batch
constexpr int NC  = Np / NCH;     // 2048 points per block
constexpr int STT = 40;
constexpr int SMEM_K3 = Cc * STT * 2 + 16 * STT * 2 + KO * 4;

__global__ __launch_bounds__(256) void k_pool(
    const float* __restrict__ ff, const float* __restrict__ voted,
    const float* __restrict__ kpp, float* __restrict__ sumexp,
    float* __restrict__ kp_raw) {
  extern __shared__ char smem[];
  _Float16* ldsT = (_Float16*)smem;          // Cc x STT  (feat transposed, f16)
  _Float16* wA   = ldsT + Cc * STT;          // 16 x STT  (exp weights, A layout src)
  float*    kps  = (float*)(wA + 16 * STT);  // 48

  const int b  = blockIdx.x / NCH;
  const int n0 = (blockIdx.x % NCH) * NC;
  const int tid = threadIdx.x;
  const int wave = tid >> 5, lane = tid & 31, l16 = lane & 15, hi = lane >> 4;

  if (tid < KO) kps[tid] = kpp[b * KO + tid];

  const int m   = tid & 15;        // keypoint this thread evaluates
  const int kk0 = (tid >> 4) * 2;  // its two points per 32-point chunk
  float partial = 0.f;

  v8f acc0 = {}, acc1 = {};
  const int c0 = wave * 32;        // this wave's 32 channels
  const float* ffb = ff + ((size_t)b * Np + n0) * Cc;
  const float* vtb = voted + ((size_t)b * Np + n0) * (Kk * 3);
  float px = 0.f, py = 0.f, pz = 0.f;

  for (int it = 0; it < NC / 32; ++it) {
    __syncthreads();  // previous iteration's WMMA reads done before overwrite
    if (it == 0) { px = kps[m * 3 + 0]; py = kps[m * 3 + 1]; pz = kps[m * 3 + 2]; }
    // stage feat chunk transposed: ldsT[c][kk]; vectorized global read
    const float* fchunk = ffb + (size_t)it * 32 * Cc;
    for (int i = tid; i < 32 * Cc / 4; i += 256) {
      const int kk = i >> 6, c4 = (i & 63) * 4;
      float4 f = ((const float4*)fchunk)[i];
      ldsT[(c4 + 0) * STT + kk] = (_Float16)f.x;
      ldsT[(c4 + 1) * STT + kk] = (_Float16)f.y;
      ldsT[(c4 + 2) * STT + kk] = (_Float16)f.z;
      ldsT[(c4 + 3) * STT + kk] = (_Float16)f.w;
    }
    // exp(-dist) tile: wA[m][kk]  (A operand for the pooling GEMM)
#pragma unroll
    for (int j = 0; j < 2; ++j) {
      int kk = kk0 + j;
      const float* vp = vtb + ((size_t)(it * 32 + kk) * Kk + m) * 3;
      float dx = vp[0] - px, dy = vp[1] - py, dz = vp[2] - pz;
      float e = __expf(-sqrtf(dx * dx + dy * dy + dz * dz));
      wA[m * STT + kk] = (_Float16)e;
      partial += e;
    }
    __syncthreads();
    v16h a  = wmma_ld_a(wA + l16 * STT, hi);
    v16h b0 = wmma_ld_b(ldsT + (size_t)(c0 + l16) * STT + hi * 16);
    v16h b1 = wmma_ld_b(ldsT + (size_t)(c0 + 16 + l16) * STT + hi * 16);
    acc0 = wmma16(a, b0, acc0);
    acc1 = wmma16(a, b1, acc1);
  }

  atomicAdd(&sumexp[b * Kk + m], partial);

  float* kb = kp_raw + (size_t)b * Kk * Cc;
#pragma unroll
  for (int r = 0; r < 8; ++r) {
    int kidx = r + hi * 8;
    atomicAdd(&kb[kidx * Cc + c0 + l16],      acc0[r]);
    atomicAdd(&kb[kidx * Cc + c0 + 16 + l16], acc1[r]);
  }
}

// ---------------- kernel 4: normalize + keypoint MLP ----------------
constexpr int SMEM_K4 = 2 * 128 * SA * 2;

__global__ __launch_bounds__(256) void k_head(
    const float* __restrict__ kp_raw, const float* __restrict__ sumexp,
    const _Float16* __restrict__ Wm1h, const _Float16* __restrict__ Wm2h,
    const _Float16* __restrict__ Wm3h,
    const float* __restrict__ sm1, const float* __restrict__ tm1,
    const float* __restrict__ sm2, const float* __restrict__ tm2,
    float* __restrict__ out) {
  extern __shared__ char smem[];
  _Float16* bufA = (_Float16*)smem;     // 128 x SA
  _Float16* bufB = bufA + 128 * SA;

  const int tid = threadIdx.x;
  const int wave = tid >> 5, lane = tid & 31, l16 = lane & 15, hi = lane >> 4;

  for (int i = tid; i < 128 * Cc / 4; i += 256) {
    const int row = i >> 6, c4 = (i & 63) * 4;
    const float inv = 1.f / sumexp[row];
    float4 f = ((const float4*)kp_raw)[i];
    v4h h = { (_Float16)(f.x * inv), (_Float16)(f.y * inv),
              (_Float16)(f.z * inv), (_Float16)(f.w * inv) };
    *(v4h*)(bufA + row * SA + c4) = h;  // kp_feat
  }
  __syncthreads();

  const int m0 = wave * 16;  // wave handles one batch's 16 keypoints
  for (int layer = 0; layer < 2; ++layer) {
    const _Float16* src = layer ? bufB : bufA;
    _Float16*       dst = layer ? bufA : bufB;
    const _Float16* Wh  = layer ? Wm2h : Wm1h;
    const float*    ss  = layer ? sm2 : sm1;
    const float*    tt  = layer ? tm2 : tm1;
    const _Float16* arow = src + (m0 + l16) * SA;
    v16h areg[8];
#pragma unroll
    for (int kc = 0; kc < 8; ++kc) areg[kc] = wmma_ld_a(arow + kc * 32, hi);
    for (int nt = 0; nt < 16; ++nt) {
      const int nc = nt * 16;
      v8f acc = {};
#pragma unroll
      for (int kc = 0; kc < 8; ++kc) {
        v16h bm = wmma_ld_b(Wh + (size_t)(nc + l16) * Cc + kc * 32 + hi * 16);
        acc = wmma16(areg[kc], bm, acc);
      }
      const int col = nc + l16;
      const float s = ss[col], t = tt[col];
#pragma unroll
      for (int r = 0; r < 8; ++r) {
        float v = fmaxf(acc[r] * s + t, 0.f);
        dst[(m0 + r + hi * 8) * SA + col] = (_Float16)v;
      }
    }
  }
  // final linear (no BN/ReLU) -> y
  const _Float16* arow = bufA + (m0 + l16) * SA;
  v16h areg[8];
#pragma unroll
  for (int kc = 0; kc < 8; ++kc) areg[kc] = wmma_ld_a(arow + kc * 32, hi);
  for (int nt = 0; nt < 16; ++nt) {
    const int nc = nt * 16;
    v8f acc = {};
#pragma unroll
    for (int kc = 0; kc < 8; ++kc) {
      v16h bm = wmma_ld_b(Wm3h + (size_t)(nc + l16) * Cc + kc * 32 + hi * 16);
      acc = wmma16(areg[kc], bm, acc);
    }
    const int col = nc + l16;
#pragma unroll
    for (int r = 0; r < 8; ++r)
      out[(size_t)(m0 + r + hi * 8) * Cc + col] = acc[r];
  }
}

// ---------------- host launcher ----------------
extern "C" void kernel_launch(void* const* d_in, const int* in_sizes, int n_in,
                              void* d_out, int out_size, void* d_ws, size_t ws_size,
                              hipStream_t stream) {
  (void)in_sizes; (void)n_in; (void)out_size; (void)ws_size;
  const float* ff  = (const float*)d_in[0];
  const float* xyz = (const float*)d_in[1];
  PrepArgs p;
  p.W1 = (const float*)d_in[2];
  p.g1 = (const float*)d_in[3];  p.b1 = (const float*)d_in[4];
  p.m1 = (const float*)d_in[5];  p.v1 = (const float*)d_in[6];
  p.W2 = (const float*)d_in[7];
  p.g2 = (const float*)d_in[8];  p.b2 = (const float*)d_in[9];
  p.m2 = (const float*)d_in[10]; p.v2 = (const float*)d_in[11];
  p.W3 = (const float*)d_in[12];
  p.g3 = (const float*)d_in[13]; p.b3 = (const float*)d_in[14];
  p.m3 = (const float*)d_in[15]; p.v3 = (const float*)d_in[16];
  p.Wo = (const float*)d_in[17];
  const float* bo = (const float*)d_in[18];
  p.Wm1 = (const float*)d_in[19];
  p.gm1 = (const float*)d_in[20]; p.bm1 = (const float*)d_in[21];
  p.mm1 = (const float*)d_in[22]; p.vm1 = (const float*)d_in[23];
  p.Wm2 = (const float*)d_in[24];
  p.gm2 = (const float*)d_in[25]; p.bm2 = (const float*)d_in[26];
  p.mm2 = (const float*)d_in[27]; p.vm2 = (const float*)d_in[28];
  p.Wm3 = (const float*)d_in[29];
  char* ws = (char*)d_ws;
  p.ws = ws;
  float* out = (float*)d_out;

  (void)hipFuncSetAttribute((const void*)k_backbone,
      hipFuncAttributeMaxDynamicSharedMemorySize, SMEM_K1);
  (void)hipFuncSetAttribute((const void*)k_pool,
      hipFuncAttributeMaxDynamicSharedMemorySize, SMEM_K3);
  (void)hipFuncSetAttribute((const void*)k_head,
      hipFuncAttributeMaxDynamicSharedMemorySize, SMEM_K4);

  k_prep<<<512, 256, 0, stream>>>(p);

  k_backbone<<<Bn * (Np / TM), 256, SMEM_K1, stream>>>(
      ff, xyz,
      (const _Float16*)(ws + OFF_W1H), (const _Float16*)(ws + OFF_W2H),
      (const _Float16*)(ws + OFF_W3H), (const _Float16*)(ws + OFF_WOH),
      (const float*)(ws + OFF_S1), (const float*)(ws + OFF_T1),
      (const float*)(ws + OFF_S2), (const float*)(ws + OFF_T2),
      (const float*)(ws + OFF_S3), (const float*)(ws + OFF_T3),
      bo,
      (float*)(ws + OFF_VOTED), (float*)(ws + OFF_VSUM));

  k_kppos<<<1, 384, 0, stream>>>((const float*)(ws + OFF_VSUM),
                                 (float*)(ws + OFF_KPPOS), out);

  k_pool<<<Bn * NCH, 256, SMEM_K3, stream>>>(
      ff, (const float*)(ws + OFF_VOTED), (const float*)(ws + OFF_KPPOS),
      (float*)(ws + OFF_SUMEXP), (float*)(ws + OFF_KPRAW));

  k_head<<<1, 256, SMEM_K4, stream>>>(
      (const float*)(ws + OFF_KPRAW), (const float*)(ws + OFF_SUMEXP),
      (const _Float16*)(ws + OFF_WM1H), (const _Float16*)(ws + OFF_WM2H),
      (const _Float16*)(ws + OFF_WM3H),
      (const float*)(ws + OFF_SM1), (const float*)(ws + OFF_TM1),
      (const float*)(ws + OFF_SM2), (const float*)(ws + OFF_TM2),
      out + 384);
}
